// FactorAtt_ConvRelPosEnc_72215580115134
// MI455X (gfx1250) — matrix operand order repaired and checked by reading, not compiled, not run on hardware
//
#include <hip/hip_runtime.h>
#include <stdint.h>

#define BDIM 8
#define SEQ  4096
#define CH   768
#define NH   12
#define HD   64
#define TC   2304              // 3*CH
#define MROWS (BDIM*SEQ)       // 32768

typedef __attribute__((ext_vector_type(16))) __bf16         bf16x16;
typedef __attribute__((ext_vector_type(8)))  float          f32x8;
typedef __attribute__((ext_vector_type(4)))  float          f32x4;
typedef __attribute__((ext_vector_type(8)))  unsigned short u16x8;
typedef __attribute__((ext_vector_type(4)))  unsigned short u16x4;

union Frag { bf16x16 bf; u16x8 h[2]; };

__device__ __forceinline__ unsigned short f2bf(float f) {
  union { float f; unsigned u; } c; c.f = f;
  unsigned u = c.u;
  u += 0x7fffu + ((u >> 16) & 1u);      // round-to-nearest-even
  return (unsigned short)(u >> 16);
}
__device__ __forceinline__ float bf2f(unsigned short h) {
  union { unsigned u; float f; } c; c.u = ((unsigned)h) << 16;
  return c.f;
}

// ---- CDNA5 async memory->LDS copy (ASYNCcnt-tracked, no VGPR landing zone).
// Used only for one-shot prologue staging (inline asm is a scheduling barrier,
// so keep it OUT of software-pipelined hot loops).
__device__ __forceinline__ void async_ld_b128(unsigned lds_addr, const void* gaddr) {
  asm volatile("global_load_async_to_lds_b128 %0, %1, off"
               :: "v"(lds_addr), "v"(gaddr) : "memory");
}
__device__ __forceinline__ void wait_async0() {
  asm volatile("s_wait_asynccnt 0x0" ::: "memory");
}
__device__ __forceinline__ unsigned lds_off(const void* p) {
  return (unsigned)(size_t)p;            // flat->LDS mapping truncates to addr[31:0]
}

// ---------------------------------------------------------------- cvt f32->bf16
__global__ void cvt_f32_bf16(const float* __restrict__ in,
                             unsigned short* __restrict__ out, int n) {
  int i = (blockIdx.x * blockDim.x + threadIdx.x) * 4;
  if (i + 3 < n) {
    f32x4 v = *(const f32x4*)(in + i);
    u16x4 o;
    o.x = f2bf(v.x); o.y = f2bf(v.y); o.z = f2bf(v.z); o.w = f2bf(v.w);
    *(u16x4*)(out + i) = o;
  } else {
    for (; i < n; ++i) out[i] = f2bf(in[i]);
  }
}

__global__ void zero_f32(float* __restrict__ p, int n) {
  int i = blockIdx.x * blockDim.x + threadIdx.x;
  if (i < n) p[i] = 0.f;
}

// ---------------------------------------------------------------- W pre-swizzle
// Wsw[nt][kb][c][l][e] = bf16(W[kb*32 + 16*(l>>4) + e][nt*128 + c*16 + (l&15)])
// i.e. W stored directly in per-lane WMMA B-fragment order. One-time pass.
__global__ __launch_bounds__(256) void w_swizzle(
    const float* __restrict__ W, unsigned short* __restrict__ Wsw) {
  int tid = blockIdx.x * blockDim.x + threadIdx.x;   // 110592 fragment rows
  if (tid >= 18 * 24 * 8 * 32) return;
  int l  = tid & 31;
  int c  = (tid >> 5) & 7;
  int kb = (tid >> 8) % 24;
  int nt = tid / (24 * 256);
  int n  = nt * 128 + c * 16 + (l & 15);
  int k0 = kb * 32 + 16 * (l >> 4);
  u16x8 o0, o1;
  #pragma unroll
  for (int e = 0; e < 8; ++e) o0[e] = f2bf(W[(size_t)(k0 + e) * TC + n]);
  #pragma unroll
  for (int e = 0; e < 8; ++e) o1[e] = f2bf(W[(size_t)(k0 + 8 + e) * TC + n]);
  *(u16x8*)(Wsw + (size_t)tid * 16)     = o0;
  *(u16x8*)(Wsw + (size_t)tid * 16 + 8) = o1;
}

// ---------------------------------------------------------------- QKV GEMM
// C[M=32768, 2304] = X[M,768] * W[768,2304], bf16 in (W pre-swizzled), bf16 out.
// Block tile 128x128, BK=32, 256 threads = 8 waves (2x4), wave tile 64x32.
// A tile staged through LDS (sync loads -> compiler software-pipelines);
// B fragments loaded straight from global (L2-resident).
__global__ __launch_bounds__(256) void qkv_gemm(
    const unsigned short* __restrict__ X,
    const unsigned short* __restrict__ Wsw,
    unsigned short* __restrict__ Q) {
  __shared__ unsigned short As[128 * 40];     // row-major, padded stride 40

  const int t    = threadIdx.x;
  const int wave = t >> 5, lane = t & 31;
  const int m0 = blockIdx.y * 128;
  const int nt = blockIdx.x;                  // 0..17
  const int wr = wave >> 2;                   // 0..1  (64-row slab)
  const int wc = wave & 3;                    // 0..3  (32-col slab)
  const int lhalf = lane >> 4;                // 0/1
  const int l16   = lane & 15;

  f32x8 acc[4][2] = {};

  for (int kb = 0; kb < 24; ++kb) {
    const int k0 = kb * 32;
    // ---- stage A tile 128x32 (row-major, pad 40)
    #pragma unroll
    for (int i = 0; i < 2; ++i) {
      int flat = (i * 256 + t) * 8;
      int r = flat >> 5, c = flat & 31;
      u16x8 v = *(const u16x8*)(X + (size_t)(m0 + r) * CH + k0 + c);
      *(u16x8*)(As + r * 40 + c) = v;
    }
    // ---- B fragments straight from pre-swizzled global W
    Frag fb[2];
    #pragma unroll
    for (int c = 0; c < 2; ++c) {
      const unsigned short* wp =
          Wsw + ((((size_t)nt * 24 + kb) * 8 + (wc * 2 + c)) * 32 + lane) * 16;
      fb[c].h[0] = *(const u16x8*)(wp);
      fb[c].h[1] = *(const u16x8*)(wp + 8);
    }
    if (kb + 1 < 24) {
      __builtin_prefetch(X + (size_t)(m0 + (t & 127)) * CH + k0 + 32 + (t >> 7) * 16, 0, 1);
    }
    __syncthreads();

    Frag fa[4];
    #pragma unroll
    for (int r = 0; r < 4; ++r) {
      const unsigned short* p = As + (wr * 64 + r * 16 + l16) * 40 + lhalf * 16;
      fa[r].h[0] = *(const u16x8*)(p);
      fa[r].h[1] = *(const u16x8*)(p + 8);
    }
    #pragma unroll
    for (int r = 0; r < 4; ++r)
      #pragma unroll
      for (int c = 0; c < 2; ++c)
        acc[r][c] = __builtin_amdgcn_wmma_f32_16x16x32_bf16(
            false, fa[r].bf, false, fb[c].bf, (short)0, acc[r][c], false, false);
    __syncthreads();
  }

  #pragma unroll
  for (int r = 0; r < 4; ++r)
    #pragma unroll
    for (int c = 0; c < 2; ++c) {
      int row0 = m0 + wr * 64 + r * 16 + lhalf * 8;
      int col  = nt * 128 + wc * 32 + c * 16 + l16;
      #pragma unroll
      for (int e = 0; e < 8; ++e)
        Q[(size_t)(row0 + e) * TC + col] = f2bf(acc[r][c][e]);
    }
}

// ---------------------------------------------------------------- softmax stats
// Per (b,h,d): running max & sum of exp over N (online softmax recurrence).
__global__ __launch_bounds__(256) void k_softmax_stats(
    const unsigned short* __restrict__ Qkv,
    float* __restrict__ kmax, float* __restrict__ ksum) {
  const int bh = blockIdx.x;
  const int b = bh / NH, h = bh % NH;
  const int t = threadIdx.x;
  const int d = t & 63, s = t >> 6;
  const unsigned short* base = Qkv + (size_t)b * SEQ * TC + CH + h * HD + d;

  float m = -3.0e38f, sum = 0.f;
  for (int n = s; n < SEQ; n += 4) {
    float v = bf2f(base[(size_t)n * TC]);
    if (v > m) { sum = sum * __expf(m - v) + 1.f; m = v; }
    else       { sum += __expf(v - m); }
  }
  __shared__ float sm[256], ss[256];
  sm[t] = m; ss[t] = sum;
  __syncthreads();
  if (s == 0) {
    float M = sm[d];
    #pragma unroll
    for (int i = 1; i < 4; ++i) M = fmaxf(M, sm[d + 64 * i]);
    float S = 0.f;
    #pragma unroll
    for (int i = 0; i < 4; ++i) S += ss[d + 64 * i] * __expf(sm[d + 64 * i] - M);
    kmax[bh * 64 + d] = M;
    ksum[bh * 64 + d] = S;
  }
}

// ---------------------------------------------------------------- kTv partial
// kTvF[bh][d][e] += sum_n exp(k[n,d]-max_d) * v[n,e], K-chunked over blockIdx.y.
__global__ __launch_bounds__(128) void ktv_partial(
    const unsigned short* __restrict__ Qkv,
    const float* __restrict__ kmax,
    float* __restrict__ kTvF) {
  const int bh = blockIdx.x;
  const int b = bh / NH, h = bh % NH;
  const int chunk = blockIdx.y;
  const int t = threadIdx.x;
  const int wave = t >> 5, lane = t & 31;
  const int lhalf = lane >> 4, l16 = lane & 15;

  __shared__ unsigned short Ae[64 * 40];      // exp(k)^T tile, [d][n], pad 40
  __shared__ unsigned short Vs[4 * 32 * 16];  // v tile swizzled
  __shared__ float smax[64];
  if (t < 64) smax[t] = kmax[bh * 64 + t];
  __syncthreads();

  f32x8 acc[4] = {};
  for (int it = 0; it < 32; ++it) {
    const int n0 = chunk * 1024 + it * 32;
    // stage exp(k)^T : read [n][d], write [d][n]
    #pragma unroll
    for (int i = 0; i < 2; ++i) {
      int flat = (i * 128 + t) * 8;
      int n = flat >> 6, dc = flat & 63;
      u16x8 v = *(const u16x8*)(Qkv + (size_t)(b * SEQ + n0 + n) * TC + CH + h * HD + dc);
      #pragma unroll
      for (int q = 0; q < 8; ++q) {
        int dd = dc + q;
        Ae[dd * 40 + n] = f2bf(__expf(bf2f(v[q]) - smax[dd]));
      }
    }
    // stage v swizzled
    #pragma unroll
    for (int i = 0; i < 2; ++i) {
      int flat = (i * 128 + t) * 8;
      int n = flat >> 6, j = flat & 63;
      u16x8 v = *(const u16x8*)(Qkv + (size_t)(b * SEQ + n0 + n) * TC + 2 * CH + h * HD + j);
      int lbase = 16 * (n >> 4), e = n & 15;
      #pragma unroll
      for (int q = 0; q < 8; ++q) {
        int jj = j + q;
        Vs[((jj >> 4) * 32 + (jj & 15) + lbase) * 16 + e] = v[q];
      }
    }
    __syncthreads();
    Frag fa, fb;
    const unsigned short* pa = Ae + (wave * 16 + l16) * 40 + lhalf * 16;
    fa.h[0] = *(const u16x8*)pa;
    fa.h[1] = *(const u16x8*)(pa + 8);
    #pragma unroll
    for (int c = 0; c < 4; ++c) {
      const unsigned short* pb = Vs + (c * 32 + lane) * 16;
      fb.h[0] = *(const u16x8*)pb;
      fb.h[1] = *(const u16x8*)(pb + 8);
      acc[c] = __builtin_amdgcn_wmma_f32_16x16x32_bf16(
          false, fa.bf, false, fb.bf, (short)0, acc[c], false, false);
    }
    __syncthreads();
  }
  #pragma unroll
  for (int c = 0; c < 4; ++c) {
    int row = wave * 16 + lhalf * 8;
    int col = c * 16 + l16;
    #pragma unroll
    for (int e = 0; e < 8; ++e)
      atomicAdd(&kTvF[(size_t)bh * 4096 + (row + e) * 64 + col], acc[c][e]);
  }
}

// ---------------------------------------------------------------- kTv finalize
__global__ void ktv_finalize(const float* __restrict__ kTvF,
                             const float* __restrict__ ksum,
                             unsigned short* __restrict__ kTvB, int n) {
  int i = blockIdx.x * blockDim.x + threadIdx.x;
  if (i < n) {
    int bh = i >> 12;
    int d  = (i >> 6) & 63;
    kTvB[i] = f2bf(kTvF[i] / ksum[bh * 64 + d]);
  }
}

// ---------------------------------------------------------------- out = q @ kTv
// 128-row tile of q per block (8 waves, each 16x64), K=64 (two 32-steps).
// q tile staged with CDNA5 async-to-LDS (one-shot prologue, then barrier).
// Writes f32 directly to d_out at the [B,H,N,D]-flat offset (== reshape).
__global__ __launch_bounds__(256) void out_gemm(
    const unsigned short* __restrict__ Qkv,
    const unsigned short* __restrict__ kTvB,
    float* __restrict__ out) {
  const int bh = blockIdx.y;
  const int b = bh / NH, h = bh % NH;
  const int n0 = blockIdx.x * 128;
  const int t = threadIdx.x;
  const int wave = t >> 5, lane = t & 31;
  const int lhalf = lane >> 4, l16 = lane & 15;

  __shared__ unsigned short Qs[128 * 72];       // q tile row-major, pad 72
  __shared__ unsigned short Kt[2 * 4 * 32 * 16];// kTv swizzled per 32-kstep

  // async-stage the q tile (contiguous 16B chunks both sides)
  #pragma unroll
  for (int i = 0; i < 4; ++i) {
    int flat = (i * 256 + t) * 8;
    int r = flat >> 6, c = flat & 63;
    async_ld_b128(lds_off(&Qs[r * 72 + c]),
                  Qkv + (size_t)(b * SEQ + n0 + r) * TC + h * HD + c);
  }
  // kTv tile needs the fragment swizzle -> sync scatter path
  #pragma unroll
  for (int i = 0; i < 2; ++i) {
    int flat = (i * 256 + t) * 8;
    int k = flat >> 6, j = flat & 63;
    u16x8 v = *(const u16x8*)(kTvB + (size_t)bh * 4096 + k * 64 + j);
    int ks = k >> 5, kk = k & 31;
    int lbase = 16 * (kk >> 4), e = kk & 15;
    #pragma unroll
    for (int q = 0; q < 8; ++q) {
      int jj = j + q;
      Kt[ks * 2048 + ((jj >> 4) * 32 + (jj & 15) + lbase) * 16 + e] = v[q];
    }
  }
  wait_async0();
  __syncthreads();

  f32x8 acc[4] = {};
  #pragma unroll
  for (int ks = 0; ks < 2; ++ks) {
    Frag fa, fb;
    const unsigned short* pa = Qs + (wave * 16 + l16) * 72 + ks * 32 + lhalf * 16;
    fa.h[0] = *(const u16x8*)pa;
    fa.h[1] = *(const u16x8*)(pa + 8);
    #pragma unroll
    for (int c = 0; c < 4; ++c) {
      const unsigned short* pb = Kt + ks * 2048 + (c * 32 + lane) * 16;
      fb.h[0] = *(const u16x8*)pb;
      fb.h[1] = *(const u16x8*)(pb + 8);
      acc[c] = __builtin_amdgcn_wmma_f32_16x16x32_bf16(
          false, fa.bf, false, fb.bf, (short)0, acc[c], false, false);
    }
  }
  #pragma unroll
  for (int c = 0; c < 4; ++c) {
    int row = wave * 16 + lhalf * 8;
    int col = c * 16 + l16;
    #pragma unroll
    for (int e = 0; e < 8; ++e)
      out[((size_t)bh * SEQ + n0 + row + e) * HD + col] = acc[c][e];
  }
}

// ---------------------------------------------------------------- launcher
extern "C" void kernel_launch(void* const* d_in, const int* in_sizes, int n_in,
                              void* d_out, int out_size, void* d_ws, size_t ws_size,
                              hipStream_t stream) {
  (void)in_sizes; (void)n_in; (void)out_size; (void)ws_size;
  const float* x = (const float*)d_in[0];
  const float* W = (const float*)d_in[1];
  float* out = (float*)d_out;
  char* ws = (char*)d_ws;

  const size_t nX = (size_t)MROWS * CH;   // 25,165,824
  const size_t nW = (size_t)CH * TC;      //  1,769,472

  unsigned short* Xb  = (unsigned short*)ws;
  unsigned short* Wsw = (unsigned short*)(ws + nX * 2);
  unsigned short* Qkv = (unsigned short*)(ws + nX * 2 + nW * 2);
  char* p = ws + nX * 2 + nW * 2 + (size_t)MROWS * TC * 2;
  float* kmaxp = (float*)p;           p += (size_t)96 * 64 * 4;
  float* ksump = (float*)p;           p += (size_t)96 * 64 * 4;
  float* kTvF  = (float*)p;           p += (size_t)96 * 4096 * 4;
  unsigned short* kTvB = (unsigned short*)p;

  cvt_f32_bf16<<<dim3((unsigned)((nX / 4 + 255) / 256)), 256, 0, stream>>>(x, Xb, (int)nX);
  w_swizzle<<<dim3((18 * 24 * 8 * 32 + 255) / 256), 256, 0, stream>>>(W, Wsw);
  qkv_gemm<<<dim3(18, 256), 256, 0, stream>>>(Xb, Wsw, Qkv);
  k_softmax_stats<<<dim3(96), 256, 0, stream>>>(Qkv, kmaxp, ksump);
  zero_f32<<<dim3((96 * 4096 + 255) / 256), 256, 0, stream>>>(kTvF, 96 * 4096);
  ktv_partial<<<dim3(96, 4), 128, 0, stream>>>(Qkv, kmaxp, kTvF);
  ktv_finalize<<<dim3((96 * 4096 + 255) / 256), 256, 0, stream>>>(kTvF, ksump, kTvB, 96 * 4096);
  out_gemm<<<dim3(32, 96), 256, 0, stream>>>(Qkv, kTvB, out);
}
